// HierarchicalSparseAttention_64261300683310
// MI455X (gfx1250) — matrix operand order repaired and checked by reading, not compiled, not run on hardware
//
#include <hip/hip_runtime.h>
#include <hip/hip_bf16.h>
#include <stdint.h>

// ---------------------------------------------------------------------------
// CDNA5 (gfx1250) wave32 WMMA types & helpers
// ---------------------------------------------------------------------------
typedef __bf16 bf16_t;
typedef __attribute__((ext_vector_type(8)))  bf16_t bf16x8;
typedef __attribute__((ext_vector_type(16))) bf16_t bf16x16;
typedef __attribute__((ext_vector_type(8)))  float  f32x8;

__device__ __forceinline__ uint16_t f2bf(float f) {
  union { float f; uint32_t u; } v; v.f = f;
  uint32_t u = v.u;
  uint32_t r = (u + 0x7FFFu + ((u >> 16) & 1u)) >> 16;   // round-to-nearest-even
  return (uint16_t)r;
}
__device__ __forceinline__ uint32_t f2bf2(float a, float b) {
  return (uint32_t)f2bf(a) | ((uint32_t)f2bf(b) << 16);
}

// Build a 16-element bf16 fragment from two 16-byte LDS/global chunks.
// A-fragments: p1 = p0 + 16 elements (K split ..+7 / +16..+23 per ISA layout)
// B-fragments: p1 = p0 + 16 elements (same K split as A)
__device__ __forceinline__ bf16x16 ld_frag(const uint16_t* p0, const uint16_t* p1) {
  bf16x8 lo = *reinterpret_cast<const bf16x8*>(p0);
  bf16x8 hi = *reinterpret_cast<const bf16x8*>(p1);
  return __builtin_shufflevector(lo, hi, 0,1,2,3,4,5,6,7,8,9,10,11,12,13,14,15);
}

__device__ __forceinline__ f32x8 wmma_bf16(bf16x16 a, bf16x16 b, f32x8 c) {
  return __builtin_amdgcn_wmma_f32_16x16x32_bf16(false, a, false, b, (short)0, c,
                                                 false, false);
}

// ---------------------------------------------------------------------------
// Static problem config (matches reference init_kwargs)
// ---------------------------------------------------------------------------
#define CFG_N    2
#define CFG_L    2048
#define CFG_E    2048
#define CFG_H    16
#define CFG_HKV  4
#define CFG_G    4        // H / HKV
#define CFG_D    128
#define CFG_S    64
#define CFG_B    32       // L / S
#define CFG_K    16       // top-K chunks

// ---------------------------------------------------------------------------
// Kernel 1: fp32 -> bf16 cast (for weight matrices)
// ---------------------------------------------------------------------------
__global__ __launch_bounds__(256) void k_cast_bf16(const float* __restrict__ src,
                                                   uint16_t* __restrict__ dst,
                                                   int n4) {
  int i = blockIdx.x * 256 + threadIdx.x;
  if (i >= n4) return;
  float4 v = reinterpret_cast<const float4*>(src)[i];
  uint2 o; o.x = f2bf2(v.x, v.y); o.y = f2bf2(v.z, v.w);
  reinterpret_cast<uint2*>(dst)[i] = o;
}

// ---------------------------------------------------------------------------
// Kernel 2: RMSNorm (one row of E=2048 per block) + bf16 cast
// ---------------------------------------------------------------------------
__global__ __launch_bounds__(256) void k_rmsnorm_cast(const float* __restrict__ x,
                                                      const float* __restrict__ w,
                                                      uint16_t* __restrict__ y) {
  const int row = blockIdx.x;
  const float* xr = x + (size_t)row * CFG_E;
  float4 v[2];
  float ss = 0.f;
#pragma unroll
  for (int i = 0; i < 2; ++i) {
    int idx = (threadIdx.x + i * 256) * 4;
    v[i] = *reinterpret_cast<const float4*>(xr + idx);
    ss += v[i].x * v[i].x + v[i].y * v[i].y + v[i].z * v[i].z + v[i].w * v[i].w;
  }
#pragma unroll
  for (int off = 16; off >= 1; off >>= 1) ss += __shfl_xor(ss, off, 32);
  __shared__ float sred[8];
  if ((threadIdx.x & 31) == 0) sred[threadIdx.x >> 5] = ss;
  __syncthreads();
  float tot = 0.f;
#pragma unroll
  for (int i = 0; i < 8; ++i) tot += sred[i];
  const float rstd = rsqrtf(tot * (1.0f / (float)CFG_E) + 1e-6f);
#pragma unroll
  for (int i = 0; i < 2; ++i) {
    int idx = (threadIdx.x + i * 256) * 4;
    float4 g = *reinterpret_cast<const float4*>(w + idx);
    uint2 o;
    o.x = f2bf2(v[i].x * rstd * g.x, v[i].y * rstd * g.y);
    o.y = f2bf2(v[i].z * rstd * g.z, v[i].w * rstd * g.w);
    *reinterpret_cast<uint2*>(y + (size_t)row * CFG_E + idx) = o;
  }
}

// ---------------------------------------------------------------------------
// Kernel 3/5: C[M,N] = A[M,K] * W[N,K]^T  (bf16 in, fp32 acc)
// 256 threads = 8 waves; 128x128 tile; wave = 32x64 subtile; K-step 64.
// Register double-buffered global->LDS pipeline; batched fragment loads.
// OUT_BF16: bf16 output (LDS-staged). Else fp32 + residual, LDS-staged
// in two 64-row halves for fully coalesced float4 output.
// ---------------------------------------------------------------------------
template <bool OUT_BF16>
__global__ __launch_bounds__(256) void k_gemm_nt(const uint16_t* __restrict__ A,
                                                 const uint16_t* __restrict__ Bw,
                                                 uint16_t* __restrict__ Cbf,
                                                 float* __restrict__ Cf,
                                                 const float* __restrict__ resid,
                                                 int M, int Nc, int Kd) {
  constexpr int LDH = 72;                 // LDS row pitch (bf16): 144B, 16B-aligned
  __shared__ uint16_t sm[2 * 128 * LDH];  // As | Bs (reused as C staging)
  uint16_t* As = sm;
  uint16_t* Bs = sm + 128 * LDH;

  const int tid    = threadIdx.x;
  const int w      = tid >> 5, lane = tid & 31;
  const int lm     = lane & 15;
  const int half8  = (lane < 16) ? 0 : 8;
  const int half16 = (lane < 16) ? 0 : 16;
  const int wm = w & 3, wn = w >> 2;

  const int tiles_n = Nc >> 7;
  const int brow0 = (blockIdx.x / tiles_n) << 7;
  const int bcol0 = (blockIdx.x % tiles_n) << 7;

  const int ldr = tid >> 3;               // staging row (0..31 step over i)
  const int ldc = (tid & 7) * 8;          // staging col (bf16 elements)

  const f32x8 fz = {};
  f32x8 acc[2][4];
#pragma unroll
  for (int i = 0; i < 2; ++i)
#pragma unroll
    for (int j = 0; j < 4; ++j) acc[i][j] = fz;

  // ---- prologue: load first K-slab into registers -------------------------
  uint4 ra[4], rb[4];
#pragma unroll
  for (int i = 0; i < 4; ++i) {
    int r = ldr + 32 * i;
    ra[i] = *reinterpret_cast<const uint4*>(A  + (size_t)(brow0 + r) * Kd + ldc);
    rb[i] = *reinterpret_cast<const uint4*>(Bw + (size_t)(bcol0 + r) * Kd + ldc);
  }

  for (int k0 = 0; k0 < Kd; k0 += 64) {
    __syncthreads();                      // previous compute done reading LDS
#pragma unroll
    for (int i = 0; i < 4; ++i) {
      int r = ldr + 32 * i;
      *reinterpret_cast<uint4*>(&As[r * LDH + ldc]) = ra[i];
      *reinterpret_cast<uint4*>(&Bs[r * LDH + ldc]) = rb[i];
    }
    __syncthreads();

    // issue next slab's global loads now; they complete under the WMMAs
    if (k0 + 64 < Kd) {
#pragma unroll
      for (int i = 0; i < 4; ++i) {
        int r = ldr + 32 * i;
        ra[i] = *reinterpret_cast<const uint4*>(A  + (size_t)(brow0 + r) * Kd + k0 + 64 + ldc);
        rb[i] = *reinterpret_cast<const uint4*>(Bw + (size_t)(bcol0 + r) * Kd + k0 + 64 + ldc);
      }
    }

#pragma unroll
    for (int kk = 0; kk < 64; kk += 32) {
      bf16x16 af[2], bfr[4];
#pragma unroll
      for (int i = 0; i < 2; ++i) {       // batch all fragment loads first
        const uint16_t* p = &As[(wm * 32 + i * 16 + lm) * LDH + kk + half8];
        af[i] = ld_frag(p, p + 16);
      }
#pragma unroll
      for (int j = 0; j < 4; ++j) {
        const uint16_t* p = &Bs[(wn * 64 + j * 16 + lm) * LDH + kk + half8];
        bfr[j] = ld_frag(p, p + 16);
      }
#pragma unroll
      for (int j = 0; j < 4; ++j)
#pragma unroll
        for (int i = 0; i < 2; ++i) acc[i][j] = wmma_bf16(af[i], bfr[j], acc[i][j]);
    }
  }

  if constexpr (OUT_BF16) {
    __syncthreads();
#pragma unroll
    for (int i = 0; i < 2; ++i)
#pragma unroll
      for (int j = 0; j < 4; ++j)
#pragma unroll
        for (int r = 0; r < 8; ++r) {
          int rr = wm * 32 + i * 16 + r + half8;
          int cc = wn * 64 + j * 16 + lm;
          sm[rr * 136 + cc] = f2bf(acc[i][j][r]);
        }
    __syncthreads();
#pragma unroll
    for (int i = 0; i < 8; ++i) {               // coalesced 128x128 bf16 store
      int id = tid + 256 * i;
      int r = id >> 4, c4 = id & 15;
      uint4 d = *reinterpret_cast<const uint4*>(&sm[r * 136 + c4 * 8]);
      *reinterpret_cast<uint4*>(Cbf + (size_t)(brow0 + r) * Nc + bcol0 + c4 * 8) = d;
    }
  } else {
    // fp32 + residual: stage 64 rows at a time in LDS, coalesced f4 output
    constexpr int LDF = 132;                    // float pitch, 16B-aligned
    float* smf = reinterpret_cast<float*>(sm);  // 64*132*4 = 33792B <= 36864B
#pragma unroll
    for (int half = 0; half < 2; ++half) {
      __syncthreads();
      if ((wm >> 1) == half) {
        int rbase = (wm & 1) * 32;
#pragma unroll
        for (int i = 0; i < 2; ++i)
#pragma unroll
          for (int j = 0; j < 4; ++j)
#pragma unroll
            for (int r = 0; r < 8; ++r)
              smf[(rbase + i * 16 + r + half8) * LDF + wn * 64 + j * 16 + lm] =
                  acc[i][j][r];
      }
      __syncthreads();
#pragma unroll
      for (int i = 0; i < 8; ++i) {             // 64 rows x 32 float4
        int id = tid + 256 * i;
        int r = id >> 5, c4 = id & 31;
        float4 v = *reinterpret_cast<const float4*>(&smf[r * LDF + c4 * 4]);
        size_t o = (size_t)(brow0 + half * 64 + r) * Nc + bcol0 + c4 * 4;
        float4 rs = *reinterpret_cast<const float4*>(resid + o);
        v.x += rs.x; v.y += rs.y; v.z += rs.z; v.w += rs.w;
        *reinterpret_cast<float4*>(Cf + o) = v;
      }
    }
  }
}

// ---------------------------------------------------------------------------
// Kernel 4: chunked attention with softmax1, one (n,b,hkv,g) per workgroup.
// 128 threads = 4 waves; wave owns 16 query rows. Per chunk: QK^T (WMMA),
// softmax1 over 64 keys, weighted PV (WMMA) accumulated over 16 chunks.
// ---------------------------------------------------------------------------
__global__ __launch_bounds__(128) void k_chunk_attn(const uint16_t* __restrict__ Qb,
                                                    const float* __restrict__ mem_k,
                                                    const float* __restrict__ mem_v,
                                                    const int* __restrict__ indices,
                                                    const float* __restrict__ wchunk,
                                                    uint16_t* __restrict__ Ctx,
                                                    int Mc) {
  __shared__ uint16_t Ks[64 * 136];   // K chunk, row-major (also ctx staging)
  __shared__ uint16_t Vt[128 * 72];   // V chunk, transposed [d][j]
  __shared__ uint16_t Ps[4 * 16 * 72];// wave-private P tiles

  const int tid    = threadIdx.x;
  const int w      = tid >> 5, lane = tid & 31;
  const int lm     = lane & 15;
  const int half8  = (lane < 16) ? 0 : 8;
  const int half16 = (lane < 16) ? 0 : 16;

  int bid = blockIdx.x;               // ((n*B + b)*HKV + h)*G + g
  const int g = bid & 3;  bid >>= 2;
  const int h = bid & 3;  bid >>= 2;
  const int b = bid & 31; bid >>= 5;
  const int n = bid;

  const int qrow0 = n * CFG_L + b * CFG_S;
  const int col0  = (h * CFG_G + g) * CFG_D;

  // Q fragments for this wave's 16 rows, all 4 K-slices (D=128), kept in regs
  bf16x16 qf[4];
  {
    const uint16_t* qp = Qb + (size_t)(qrow0 + w * 16 + lm) * CFG_E + col0;
#pragma unroll
    for (int kk = 0; kk < 4; ++kk) {
      const uint16_t* p = qp + kk * 32 + half8;
      qf[kk] = ld_frag(p, p + 16);
    }
  }

  const f32x8 fz = {};
  f32x8 cacc[8];
#pragma unroll
  for (int i = 0; i < 8; ++i) cacc[i] = fz;

  const int    ibase  = ((n * CFG_B + b) * CFG_HKV + h) * CFG_K;
  const size_t kvbase = (size_t)n * Mc * CFG_S * CFG_HKV * CFG_D + (size_t)h * CFG_D;
  const int    rowstr = CFG_HKV * CFG_D;      // 512 floats between chunk rows

  for (int c = 0; c < CFG_K; ++c) {
    const int   idx = indices[ibase + c];
    const float wc  = wchunk[ibase + c];
    const float* kp = mem_k + kvbase + (size_t)idx * CFG_S * rowstr;
    const float* vp = mem_v + kvbase + (size_t)idx * CFG_S * rowstr;

    __syncthreads();                          // protect LDS from prior reads

    // Stage K (64x128 bf16 row-major): 2048 float4, 16 per thread
#pragma unroll
    for (int i = 0; i < 16; ++i) {
      int id = tid + 128 * i;
      int j = id >> 5, c4 = id & 31;
      float4 kd = *reinterpret_cast<const float4*>(kp + (size_t)j * rowstr + c4 * 4);
      uint2 o; o.x = f2bf2(kd.x, kd.y); o.y = f2bf2(kd.z, kd.w);
      *reinterpret_cast<uint2*>(&Ks[j * 136 + c4 * 4]) = o;
    }
    // Stage V transposed via 4x4 register-transposed blocks (packed stores)
#pragma unroll
    for (int i = 0; i < 4; ++i) {
      int id = tid + 128 * i;                 // 512 blocks of 4j x 4d
      int j0 = (id >> 5) * 4, d0 = (id & 31) * 4;
      const float* vb = vp + (size_t)j0 * rowstr + d0;
      float4 r0 = *reinterpret_cast<const float4*>(vb);
      float4 r1 = *reinterpret_cast<const float4*>(vb + rowstr);
      float4 r2 = *reinterpret_cast<const float4*>(vb + 2 * rowstr);
      float4 r3 = *reinterpret_cast<const float4*>(vb + 3 * rowstr);
      uint2 o;
      o.x = f2bf2(r0.x, r1.x); o.y = f2bf2(r2.x, r3.x);
      *reinterpret_cast<uint2*>(&Vt[(d0 + 0) * 72 + j0]) = o;
      o.x = f2bf2(r0.y, r1.y); o.y = f2bf2(r2.y, r3.y);
      *reinterpret_cast<uint2*>(&Vt[(d0 + 1) * 72 + j0]) = o;
      o.x = f2bf2(r0.z, r1.z); o.y = f2bf2(r2.z, r3.z);
      *reinterpret_cast<uint2*>(&Vt[(d0 + 2) * 72 + j0]) = o;
      o.x = f2bf2(r0.w, r1.w); o.y = f2bf2(r2.w, r3.w);
      *reinterpret_cast<uint2*>(&Vt[(d0 + 3) * 72 + j0]) = o;
    }

    // Prefetch next gathered chunk (index-dependent: HW can't predict it)
    if (c + 1 < CFG_K) {
      const int idx2 = indices[ibase + c + 1];
      const char* nk = (const char*)(mem_k + kvbase + (size_t)idx2 * CFG_S * rowstr);
      const char* nv = (const char*)(mem_v + kvbase + (size_t)idx2 * CFG_S * rowstr);
      int j = tid >> 1, part = tid & 1;
      __builtin_prefetch(nk + (size_t)j * (rowstr * 4) + part * 256, 0, 0);
      __builtin_prefetch(nv + (size_t)j * (rowstr * 4) + part * 256, 0, 0);
    }

    __syncthreads();

    // Scores: 16 (wave rows) x 64 keys; batch B-fragments per tile
    f32x8 s[4];
#pragma unroll
    for (int t = 0; t < 4; ++t) {
      bf16x16 kb[4];
#pragma unroll
      for (int kk = 0; kk < 4; ++kk) {
        const uint16_t* p = &Ks[(t * 16 + lm) * 136 + kk * 32 + half8];
        kb[kk] = ld_frag(p, p + 16);
      }
      f32x8 a = fz;
#pragma unroll
      for (int kk = 0; kk < 4; ++kk) a = wmma_bf16(qf[kk], kb[kk], a);
      s[t] = a * 0.08838834764831845f;        // D^-0.5
    }

    // softmax1 per row over 64 keys (row = 16-lane group in C layout)
#pragma unroll
    for (int r = 0; r < 8; ++r) {
      float m = fmaxf(fmaxf(s[0][r], s[1][r]), fmaxf(s[2][r], s[3][r]));
#pragma unroll
      for (int off = 1; off <= 8; off <<= 1) m = fmaxf(m, __shfl_xor(m, off, 32));
      m = fmaxf(m, 0.0f);                     // softmax1 clamp
      float e0 = __expf(s[0][r] - m), e1 = __expf(s[1][r] - m);
      float e2 = __expf(s[2][r] - m), e3 = __expf(s[3][r] - m);
      float sum = e0 + e1 + e2 + e3;
#pragma unroll
      for (int off = 1; off <= 8; off <<= 1) sum += __shfl_xor(sum, off, 32);
      float inv = wc / (sum + __expf(-m));    // fold chunk weight into P
      s[0][r] = e0 * inv; s[1][r] = e1 * inv; s[2][r] = e2 * inv; s[3][r] = e3 * inv;
    }

    // P -> wave-private LDS (C-layout scatter => A-layout readable)
    uint16_t* pw = &Ps[w * 16 * 72];
#pragma unroll
    for (int t = 0; t < 4; ++t)
#pragma unroll
      for (int r = 0; r < 8; ++r)
        pw[(r + half8) * 72 + t * 16 + lm] = f2bf(s[t][r]);

    asm volatile("s_wait_dscnt 0x0" ::: "memory");  // same-wave LDS RAW

    bf16x16 pf[2];
#pragma unroll
    for (int kk = 0; kk < 2; ++kk) {
      const uint16_t* p = &pw[lm * 72 + kk * 32 + half8];
      pf[kk] = ld_frag(p, p + 16);
    }
    // ctx(16x128) += P(16x64) @ V(64x128); batch the two V frags per d-block
#pragma unroll
    for (int nd = 0; nd < 8; ++nd) {
      const uint16_t* pv = &Vt[(nd * 16 + lm) * 72 + half8];
      bf16x16 vb0 = ld_frag(pv, pv + 16);
      bf16x16 vb1 = ld_frag(pv + 32, pv + 48);
      cacc[nd] = wmma_bf16(pf[0], vb0, cacc[nd]);
      cacc[nd] = wmma_bf16(pf[1], vb1, cacc[nd]);
    }
  }

  // Epilogue: stage ctx in LDS (reuse Ks), then coalesced bf16 store
  __syncthreads();
#pragma unroll
  for (int nd = 0; nd < 8; ++nd)
#pragma unroll
    for (int r = 0; r < 8; ++r)
      Ks[(w * 16 + r + half8) * 136 + nd * 16 + lm] = f2bf(cacc[nd][r]);
  __syncthreads();
#pragma unroll
  for (int i = 0; i < 8; ++i) {
    int id = tid + 128 * i;                   // 64 rows x 16 uint4
    int r = id >> 4, c4 = id & 15;
    uint4 d = *reinterpret_cast<const uint4*>(&Ks[r * 136 + c4 * 8]);
    *reinterpret_cast<uint4*>(Ctx + (size_t)(qrow0 + r) * CFG_E + col0 + c4 * 8) = d;
  }
}

// ---------------------------------------------------------------------------
// Launch
// ---------------------------------------------------------------------------
extern "C" void kernel_launch(void* const* d_in, const int* in_sizes, int n_in,
                              void* d_out, int out_size, void* d_ws, size_t ws_size,
                              hipStream_t stream) {
  const float* hidden  = (const float*)d_in[0];
  const float* weights = (const float*)d_in[1];
  const float* mem_k   = (const float*)d_in[2];
  const float* mem_v   = (const float*)d_in[3];
  const int*   indices = (const int*)d_in[4];
  const float* pre_w   = (const float*)d_in[5];
  const float* q_w     = (const float*)d_in[6];
  const float* o_w     = (const float*)d_in[7];
  float*       out     = (float*)d_out;

  const int Mrows = CFG_N * CFG_L;                                   // 4096
  const int Mc    = in_sizes[2] / (CFG_N * CFG_S * CFG_HKV * CFG_D); // 512

  uint16_t* xbf  = (uint16_t*)d_ws;                 // normed x    (bf16)
  uint16_t* qbf  = xbf  + (size_t)Mrows * CFG_E;    // q           (bf16)
  uint16_t* cbf  = qbf  + (size_t)Mrows * CFG_E;    // ctx         (bf16)
  uint16_t* wqbf = cbf  + (size_t)Mrows * CFG_E;    // Wq          (bf16)
  uint16_t* wobf = wqbf + (size_t)CFG_E * CFG_E;    // Wo          (bf16)

  const int n4 = CFG_E * CFG_E / 4;
  k_cast_bf16<<<(n4 + 255) / 256, 256, 0, stream>>>(q_w, wqbf, n4);
  k_cast_bf16<<<(n4 + 255) / 256, 256, 0, stream>>>(o_w, wobf, n4);
  k_rmsnorm_cast<<<Mrows, 256, 0, stream>>>(hidden, pre_w, xbf);

  dim3 ggrid((Mrows / 128) * (CFG_E / 128));        // 512 tiles
  k_gemm_nt<true><<<ggrid, 256, 0, stream>>>(xbf, wqbf, qbf, nullptr, nullptr,
                                             Mrows, CFG_E, CFG_E);

  k_chunk_attn<<<CFG_N * CFG_B * CFG_HKV * CFG_G, 128, 0, stream>>>(
      qbf, mem_k, mem_v, indices, weights, cbf, Mc);

  k_gemm_nt<false><<<ggrid, 256, 0, stream>>>(cbf, wobf, nullptr, out, hidden,
                                              Mrows, CFG_E, CFG_E);
}
